// PointTransformerVFE_85959475462562
// MI455X (gfx1250) — matrix operand order repaired and checked by reading, not compiled, not run on hardware
//
#include <hip/hip_runtime.h>

typedef __attribute__((ext_vector_type(16))) _Float16 v16h;
typedef __attribute__((ext_vector_type(8)))  float    v8f;

#define CH   64
#define NSNB 16
#define EPSL 1e-5f

// ---------------------------------------------------------------------------
// wave32 reductions
// ---------------------------------------------------------------------------
__device__ __forceinline__ float waveSum32(float v) {
#pragma unroll
  for (int m = 16; m >= 1; m >>= 1) v += __shfl_xor(v, m, 32);
  return v;
}
__device__ __forceinline__ float grpSum16(float v) {
#pragma unroll
  for (int m = 8; m >= 1; m >>= 1) v += __shfl_xor(v, m, 32);
  return v;
}
__device__ __forceinline__ float grpMax16(float v) {
#pragma unroll
  for (int m = 8; m >= 1; m >>= 1) v = fmaxf(v, __shfl_xor(v, m, 32));
  return v;
}

// ---------------------------------------------------------------------------
// Kernel 1: fused Q/K/V projection, f16 WMMA with f32 accumulate.
// One wave per 16-row tile of x; x tile read once, reused for all 3 weights.
// A-frag (16x32 f16): lane l -> row l%16, elem e -> k = e + (e>=8?8:0) + 8*(l/16)
// B-frag (32x16 f16): lane l -> col l%16, elem e -> k as in A-frag (+32 for frag1)
// C-frag (16x16 f32): lane l -> col l%16, vgpr v -> row v + 8*(l/16)
// ---------------------------------------------------------------------------
__global__ __launch_bounds__(256) void qkv_gemm(
    const float* __restrict__ x,
    const float* __restrict__ Wq, const float* __restrict__ bq,
    const float* __restrict__ Wk, const float* __restrict__ bk,
    const float* __restrict__ Wv, const float* __restrict__ bv,
    float* __restrict__ q, float* __restrict__ kf, float* __restrict__ vf,
    int n)
{
  const int lid  = threadIdx.x & 31;
  const int wid  = threadIdx.x >> 5;
  const int tile = blockIdx.x * 8 + wid;
  const int ntiles = (n + 15) >> 4;
  if (tile >= ntiles) return;                     // wave-uniform guard

  const int half = lid >> 4;
  const int m    = lid & 15;
  const bool fulltile = (tile * 16 + 16) <= n;    // wave-uniform: fast epilogue

  int row = tile * 16 + m;
  if (row >= n) row = n - 1;                      // branch-free clamp for loads

  v16h a0, a1;
#pragma unroll
  for (int e = 0; e < 16; ++e) {
    const int k = e + (e >= 8 ? 8 : 0) + 8 * half;
    a0[e] = (_Float16)x[row * CH + k];
    a1[e] = (_Float16)x[row * CH + k + 32];
  }

  const float* Ws[3]  = {Wq, Wk, Wv};
  const float* bs[3]  = {bq, bk, bv};
  float*       out[3] = {q, kf, vf};

#pragma unroll
  for (int s = 0; s < 3; ++s) {
    const float* W = Ws[s];
    const float* b = bs[s];
    float*       o = out[s];
#pragma unroll
    for (int nt = 0; nt < 4; ++nt) {
      v16h b0, b1;
#pragma unroll
      for (int e = 0; e < 16; ++e) {
        b0[e] = (_Float16)W[(e + (e >= 8 ? 8 : 0) + 8 * half) * CH + nt * 16 + m];
        b1[e] = (_Float16)W[(e + (e >= 8 ? 8 : 0) + 8 * half + 32) * CH + nt * 16 + m];
      }
      const float bias = b[nt * 16 + m];          // output column = nt*16 + (lid&15)
      v8f c;
#pragma unroll
      for (int v = 0; v < 8; ++v) c[v] = bias;
      c = __builtin_amdgcn_wmma_f32_16x16x32_f16(false, a0, false, b0, (short)0, c, false, false);
      c = __builtin_amdgcn_wmma_f32_16x16x32_f16(false, a1, false, b1, (short)0, c, false, false);

      // epilogue: lane holds column (nt*16+m), rows v + 8*half, row-stride CH
      float* obase = o + (size_t)(tile * 16 + 8 * half) * CH + nt * 16 + m;
      if (fulltile) {
#pragma unroll
        for (int v = 0; v < 8; ++v) obase[v * CH] = c[v];
      } else {                                    // cold path: partial last tile
#pragma unroll
        for (int v = 0; v < 8; ++v) {
          const int r = tile * 16 + v + 8 * half;
          if (r < n) o[r * CH + nt * 16 + m] = c[v];
        }
      }
    }
  }
}

// ---------------------------------------------------------------------------
// Kernel 2: one wave32 per point. Fuses gather + positional MLP + LN(64) +
// (16x64)@(64x8) WMMA + LN(8) + 8x8 GEMM + softmax(16) + weighted sum.
// ---------------------------------------------------------------------------
__global__ __launch_bounds__(256) void attn(
    const float* __restrict__ p,   const int* __restrict__ idx,
    const float* __restrict__ q,   const float* __restrict__ kf,
    const float* __restrict__ vf,
    const float* __restrict__ Wp1, const float* __restrict__ bp1,
    const float* __restrict__ g1,  const float* __restrict__ be1,
    const float* __restrict__ Wp2, const float* __restrict__ bp2,
    const float* __restrict__ g2,  const float* __restrict__ be2,
    const float* __restrict__ Ww1, const float* __restrict__ bw1,
    const float* __restrict__ g3,  const float* __restrict__ be3,
    const float* __restrict__ Ww2, const float* __restrict__ bw2,
    float* __restrict__ out, int n)
{
  __shared__ _Float16 aSm[8][NSNB][CH];   // post-LN relu activations (f16)
  __shared__ float  vpeSm[8][NSNB][CH];   // v + positional encoding
  __shared__ float    uSm[8][NSNB][8];    // first weight-MLP output
  __shared__ float    wSm[8][NSNB][8];    // softmaxed attention weights

  const int lid  = threadIdx.x & 31;
  const int wid  = threadIdx.x >> 5;
  const int i    = blockIdx.x * 8 + wid;
  const bool act = (i < n);                       // wave-uniform
  const int half = lid >> 4;
  const int c0 = lid, c1 = lid + 32;

  if (act) {
    // per-point / per-lane constants
    const float xq0 = q[i * CH + c0], xq1 = q[i * CH + c1];
    const float px = p[i * 3 + 0], py = p[i * 3 + 1], pz = p[i * 3 + 2];
    float wp1[9], vb1[3], vg1[3], ve1[3];
#pragma unroll
    for (int j = 0; j < 9; ++j) wp1[j] = Wp1[j];
#pragma unroll
    for (int j = 0; j < 3; ++j) { vb1[j] = bp1[j]; vg1[j] = g1[j]; ve1[j] = be1[j]; }
    const float w2a0 = Wp2[0 * CH + c0], w2b0 = Wp2[1 * CH + c0], w2c0 = Wp2[2 * CH + c0];
    const float w2a1 = Wp2[0 * CH + c1], w2b1 = Wp2[1 * CH + c1], w2c1 = Wp2[2 * CH + c1];
    const float bp2c0 = bp2[c0], bp2c1 = bp2[c1];
    const float g2c0 = g2[c0], g2c1 = g2[c1], be2c0 = be2[c0], be2c1 = be2[c1];

    // ---- Phase 1: lane<->channel; build activations and v+pe in LDS ----
    for (int t = 0; t < NSNB; ++t) {
      const int nb = idx[i * NSNB + t];
      const float prx = p[nb * 3 + 0] - px;
      const float pry = p[nb * 3 + 1] - py;
      const float prz = p[nb * 3 + 2] - pz;
      // Linear(3,3)
      float h0 = prx * wp1[0] + pry * wp1[3] + prz * wp1[6] + vb1[0];
      float h1 = prx * wp1[1] + pry * wp1[4] + prz * wp1[7] + vb1[1];
      float h2 = prx * wp1[2] + pry * wp1[5] + prz * wp1[8] + vb1[2];
      // LN(3) + relu (redundant per lane, trivial cost)
      const float mh = (h0 + h1 + h2) * (1.0f / 3.0f);
      const float dh0 = h0 - mh, dh1 = h1 - mh, dh2 = h2 - mh;
      const float rsh = rsqrtf((dh0 * dh0 + dh1 * dh1 + dh2 * dh2) * (1.0f / 3.0f) + EPSL);
      h0 = fmaxf(dh0 * rsh * vg1[0] + ve1[0], 0.0f);
      h1 = fmaxf(dh1 * rsh * vg1[1] + ve1[1], 0.0f);
      h2 = fmaxf(dh2 * rsh * vg1[2] + ve1[2], 0.0f);
      // Linear(3,64): 2 channels per lane
      const float pe0 = h0 * w2a0 + h1 * w2b0 + h2 * w2c0 + bp2c0;
      const float pe1 = h0 * w2a1 + h1 * w2b1 + h2 * w2c1 + bp2c1;
      // r_qk and LN(64) via wave reduction
      const float r0 = kf[nb * CH + c0] - xq0 + pe0;
      const float r1 = kf[nb * CH + c1] - xq1 + pe1;
      const float mean = waveSum32(r0 + r1) * (1.0f / 64.0f);
      const float d0 = r0 - mean, d1 = r1 - mean;
      const float rsv = rsqrtf(waveSum32(d0 * d0 + d1 * d1) * (1.0f / 64.0f) + EPSL);
      aSm[wid][t][c0] = (_Float16)fmaxf(d0 * rsv * g2c0 + be2c0, 0.0f);
      aSm[wid][t][c1] = (_Float16)fmaxf(d1 * rsv * g2c1 + be2c1, 0.0f);
      vpeSm[wid][t][c0] = vf[nb * CH + c0] + pe0;
      vpeSm[wid][t][c1] = vf[nb * CH + c1] + pe1;
    }
  }
  __syncthreads();

  if (act) {
    // ---- Phase 2: u = a(16x64) @ Ww1(64x8, zero-padded to 16 cols) ----
    const int mrow = lid & 15;
    v16h af0, af1, bf0, bf1;
#pragma unroll
    for (int e = 0; e < 16; ++e) {
      const int k = e + (e >= 8 ? 8 : 0) + 8 * half;
      af0[e] = aSm[wid][mrow][k];
      af1[e] = aSm[wid][mrow][k + 32];
      bf0[e] = (mrow < 8) ? (_Float16)Ww1[k * 8 + mrow]        : (_Float16)0.0f;
      bf1[e] = (mrow < 8) ? (_Float16)Ww1[(k + 32) * 8 + mrow] : (_Float16)0.0f;
    }
    v8f c;
    const float biasw = (mrow < 8) ? bw1[mrow] : 0.0f;
#pragma unroll
    for (int v = 0; v < 8; ++v) c[v] = biasw;
    c = __builtin_amdgcn_wmma_f32_16x16x32_f16(false, af0, false, bf0, (short)0, c, false, false);
    c = __builtin_amdgcn_wmma_f32_16x16x32_f16(false, af1, false, bf1, (short)0, c, false, false);
    if (mrow < 8) {
#pragma unroll
      for (int v = 0; v < 8; ++v) uSm[wid][v + 8 * half][mrow] = c[v];
    }
  }
  __syncthreads();

  if (act) {
    // ---- Phase 3: lane<->neighbor; LN(8) + 8x8 GEMM + softmax over 16 ----
    const int t = lid & 15;                       // lanes 16..31 duplicate
    float y[8];
#pragma unroll
    for (int j = 0; j < 8; ++j) y[j] = uSm[wid][t][j];
    float s = 0.0f;
#pragma unroll
    for (int j = 0; j < 8; ++j) s += y[j];
    const float mean = s * 0.125f;
    float vv = 0.0f;
#pragma unroll
    for (int j = 0; j < 8; ++j) { const float d = y[j] - mean; vv += d * d; }
    const float rsv = rsqrtf(vv * 0.125f + EPSL);
#pragma unroll
    for (int j = 0; j < 8; ++j)
      y[j] = fmaxf((y[j] - mean) * rsv * g3[j] + be3[j], 0.0f);
    float w2[8];
#pragma unroll
    for (int j2 = 0; j2 < 8; ++j2) {
      float acc = bw2[j2];
#pragma unroll
      for (int j = 0; j < 8; ++j) acc += y[j] * Ww2[j * 8 + j2];
      w2[j2] = acc;
    }
    // softmax over neighbors (groups of 16 lanes)
#pragma unroll
    for (int j2 = 0; j2 < 8; ++j2) {
      const float mx = grpMax16(w2[j2]);
      const float ex = __expf(w2[j2] - mx);
      w2[j2] = ex / grpSum16(ex);
    }
    if (lid < 16) {
#pragma unroll
      for (int j2 = 0; j2 < 8; ++j2) wSm[wid][t][j2] = w2[j2];
    }
  }
  __syncthreads();

  if (act) {
    // ---- Phase 4: lane<->channel; out[c] = sum_t vpe[t][c] * w[t][c%8] ----
    const int j = lid & 7;                        // (c0%8) == (c1%8)
    float o0 = 0.0f, o1 = 0.0f;
#pragma unroll
    for (int t = 0; t < NSNB; ++t) {
      const float wt = wSm[wid][t][j];
      o0 += vpeSm[wid][t][c0] * wt;
      o1 += vpeSm[wid][t][c1] * wt;
    }
    out[i * CH + c0] = o0;
    out[i * CH + c1] = o1;
  }
}

// ---------------------------------------------------------------------------
extern "C" void kernel_launch(void* const* d_in, const int* in_sizes, int n_in,
                              void* d_out, int out_size, void* d_ws, size_t ws_size,
                              hipStream_t stream) {
  const float* p   = (const float*)d_in[0];
  const float* x   = (const float*)d_in[1];
  const int*   idx = (const int*)  d_in[2];
  const float* Wq  = (const float*)d_in[3];
  const float* bq  = (const float*)d_in[4];
  const float* Wk  = (const float*)d_in[5];
  const float* bk  = (const float*)d_in[6];
  const float* Wv  = (const float*)d_in[7];
  const float* bv  = (const float*)d_in[8];
  const float* Wp1 = (const float*)d_in[9];
  const float* bp1 = (const float*)d_in[10];
  const float* g1  = (const float*)d_in[11];
  const float* be1 = (const float*)d_in[12];
  const float* Wp2 = (const float*)d_in[13];
  const float* bp2 = (const float*)d_in[14];
  const float* g2  = (const float*)d_in[15];
  const float* be2 = (const float*)d_in[16];
  const float* Ww1 = (const float*)d_in[17];
  const float* bw1 = (const float*)d_in[18];
  const float* g3  = (const float*)d_in[19];
  const float* be3 = (const float*)d_in[20];
  const float* Ww2 = (const float*)d_in[21];
  const float* bw2 = (const float*)d_in[22];

  const int n = in_sizes[1] / CH;                 // 100000

  float* qbuf = (float*)d_ws;
  float* kbuf = qbuf + (size_t)n * CH;
  float* vbuf = kbuf + (size_t)n * CH;

  const int ntiles  = (n + 15) / 16;
  const int gBlocks = (ntiles + 7) / 8;
  qkv_gemm<<<gBlocks, 256, 0, stream>>>(x, Wq, bq, Wk, bk, Wv, bv,
                                        qbuf, kbuf, vbuf, n);

  const int aBlocks = (n + 7) / 8;
  attn<<<aBlocks, 256, 0, stream>>>(p, idx, qbuf, kbuf, vbuf,
                                    Wp1, bp1, g1, be1, Wp2, bp2, g2, be2,
                                    Ww1, bw1, g3, be3, Ww2, bw2,
                                    (float*)d_out, n);
}
